// LSTM_19499151524089
// MI455X (gfx1250) — compile-verified, matrix operand order checked by reading, not run on hardware
//
#include <hip/hip_runtime.h>
#include <hip/hip_bf16.h>

typedef unsigned int  u32;
typedef unsigned short u16;
typedef __attribute__((ext_vector_type(16))) __bf16 v16bf;
typedef __attribute__((ext_vector_type(8)))  float  v8f;
typedef __attribute__((ext_vector_type(4)))  u32    u32x4;

#define T_STEPS 512
#define BATCH   64
#define IDIM    512
#define HDIM    1024
#define NBLK    32
#define BLKTHR  128   // 4 wave32 per block -> 128 waves total

// ---------- helpers ----------
__device__ __forceinline__ u16 f2bf(float f) {
  u32 u = __builtin_bit_cast(u32, f);
  u32 r = 0x7FFFu + ((u >> 16) & 1u);      // round-to-nearest-even
  return (u16)((u + r) >> 16);
}

// Load a 16x32 bf16 A-fragment (or 32x16 B-fragment) in the CDNA5 wave32 layout:
// lanes 0-15 hold K=[0..7] and K=[16..23]; lanes 16-31 hold K=[8..15] and K=[24..31].
// Caller passes p already offset by row*stride + k0 + khalf; we grab two 16B chunks.
__device__ __forceinline__ v16bf load_frag(const u16* p) {
  union { struct { u32x4 lo, hi; } s; v16bf v; } u;
  u.s.lo = *(const u32x4*)(p);        // K = khalf .. khalf+7
  u.s.hi = *(const u32x4*)(p + 16);   // K = 16+khalf .. 16+khalf+7
  return u.v;
}

__device__ __forceinline__ v8f wmma_bf16(v16bf a, v16bf b, v8f c) {
  return __builtin_amdgcn_wmma_f32_16x16x32_bf16(false, a, false, b, (short)0, c,
                                                 false, false);
}

__device__ __forceinline__ float sigmoid_f(float x) { return 1.0f / (1.0f + __expf(-x)); }
__device__ __forceinline__ float tanh_f(float x)    { return 1.0f - 2.0f / (__expf(2.0f * x) + 1.0f); }

// ---------- prep kernels ----------
__global__ void k_cvt_bf16(const float* __restrict__ src, u16* __restrict__ dst, int n) {
  int i = blockIdx.x * blockDim.x + threadIdx.x;
  if (i < n) dst[i] = f2bf(src[i]);
}

// dst[n][k] = bf16( W_gate[k][n&1023] ), n in [0,4096), gate order i,f,g,o
__global__ void k_pack_wT(const float* __restrict__ g0, const float* __restrict__ g1,
                          const float* __restrict__ g2, const float* __restrict__ g3,
                          u16* __restrict__ dst, int K) {
  int idx = blockIdx.x * blockDim.x + threadIdx.x;   // idx = n*K + k
  if (idx >= 4096 * K) return;
  int n = idx / K, k = idx - n * K;
  int gate = n >> 10, n0 = n & 1023;
  const float* W = (gate == 0) ? g0 : (gate == 1) ? g1 : (gate == 2) ? g2 : g3;
  dst[idx] = f2bf(W[k * 1024 + n0]);
}

__global__ void k_pack_bias(const float* __restrict__ bi, const float* __restrict__ bf_,
                            const float* __restrict__ bg, const float* __restrict__ bo,
                            float* __restrict__ dst) {
  int i = blockIdx.x * blockDim.x + threadIdx.x;
  if (i < 4096) {
    int g = i >> 10, n = i & 1023;
    dst[i] = (g == 0) ? bi[n] : (g == 1) ? bf_[n] : (g == 2) ? bg[n] : bo[n];
  }
}

__global__ void k_zero_cnt(u32* p) { if (threadIdx.x == 0) *p = 0u; }

// ---------- persistent LSTM scan ----------
__global__ __launch_bounds__(BLKTHR, 1)
void k_lstm_scan(const u16* __restrict__ xb,   // [T*B, I] bf16
                 const u16* __restrict__ wxt,  // [4096, I] bf16 (col-major weights)
                 const u16* __restrict__ wht,  // [4096, H] bf16
                 const float* __restrict__ bias,   // [4096]
                 const float* __restrict__ c0,     // [B, H] f32
                 u16* hbuf0, u16* hbuf1,           // [B, H] bf16 ping-pong
                 float* __restrict__ out,          // [T, B, H] f32
                 u32* cnt) {
  const int lane = threadIdx.x & 31;
  const int lw   = threadIdx.x >> 5;            // 0..3 local wave
  // Co-locate cache sharers: each block covers 2 strips x 2 batch-halves, so both
  // waves reading a given B column (and both reading a given A row) live in the
  // same WGP and hit the WGP cache instead of fetching twice from L2.
  const int strip = blockIdx.x * 2 + (lw & 1);  // 0..63 -> which 16 h-columns
  const int mpart = lw >> 1;                    // 0..1  -> batch rows 0-31 / 32-63
  const int colBase = strip * 16;
  const int nlo  = lane & 15;
  const int hi8  = (lane >> 4) * 8;    // serves as both khalf and C-row offset
  const int col  = colBase + nlo;
  const int rowA0 = mpart * 32 + nlo;  // A-fragment row for tile 0 of this wave

  float bb[4];
  #pragma unroll
  for (int g = 0; g < 4; ++g) bb[g] = bias[g * HDIM + col];

  // cell state kept in registers for all 512 steps, C-fragment layout
  v8f cfrag[2];
  #pragma unroll
  for (int mt = 0; mt < 2; ++mt)
    #pragma unroll
    for (int r = 0; r < 8; ++r)
      cfrag[mt][r] = c0[(mpart * 32 + mt * 16 + hi8 + r) * HDIM + col];

  u16* hb[2] = { hbuf0, hbuf1 };

  for (int t = 0; t < T_STEPS; ++t) {
    const u16* hcur  = hb[t & 1];
    u16*       hnext = hb[(t + 1) & 1];

    v8f acc[4][2];
    #pragma unroll
    for (int g = 0; g < 4; ++g)
      #pragma unroll
      for (int mt = 0; mt < 2; ++mt)
        acc[g][mt] = (v8f)(0.0f);

    // ---- x_t @ Wx : K = IDIM ----
    const u16* xrow = xb + (size_t)t * BATCH * IDIM;
    for (int k0 = 0; k0 < IDIM; k0 += 32) {
      v16bf a[2];
      #pragma unroll
      for (int mt = 0; mt < 2; ++mt)
        a[mt] = load_frag(xrow + (size_t)(rowA0 + mt * 16) * IDIM + k0 + hi8);
      #pragma unroll
      for (int g = 0; g < 4; ++g) {
        v16bf bm = load_frag(wxt + (size_t)(g * HDIM + col) * IDIM + k0 + hi8);
        #pragma unroll
        for (int mt = 0; mt < 2; ++mt)
          acc[g][mt] = wmma_bf16(a[mt], bm, acc[g][mt]);
      }
    }
    // ---- h_t @ Wh : K = HDIM ----
    for (int k0 = 0; k0 < HDIM; k0 += 32) {
      v16bf a[2];
      #pragma unroll
      for (int mt = 0; mt < 2; ++mt)
        a[mt] = load_frag(hcur + (size_t)(rowA0 + mt * 16) * HDIM + k0 + hi8);
      #pragma unroll
      for (int g = 0; g < 4; ++g) {
        v16bf bm = load_frag(wht + (size_t)(g * HDIM + col) * HDIM + k0 + hi8);
        #pragma unroll
        for (int mt = 0; mt < 2; ++mt)
          acc[g][mt] = wmma_bf16(a[mt], bm, acc[g][mt]);
      }
    }

    // ---- gates, state update, writes ----
    #pragma unroll
    for (int mt = 0; mt < 2; ++mt) {
      #pragma unroll
      for (int r = 0; r < 8; ++r) {
        float gi = sigmoid_f(acc[0][mt][r] + bb[0]);
        float gf = sigmoid_f(acc[1][mt][r] + bb[1]);
        float gg = tanh_f   (acc[2][mt][r] + bb[2]);
        float go = sigmoid_f(acc[3][mt][r] + bb[3]);
        float c  = gf * cfrag[mt][r] + gi * gg;
        cfrag[mt][r] = c;
        float h  = go * tanh_f(c);
        int row = mpart * 32 + mt * 16 + hi8 + r;
        out[((size_t)t * BATCH + row) * HDIM + col] = h;
        hnext[row * HDIM + col] = f2bf(h);
      }
    }

    // ---- grid-wide barrier (monotonic counter, release/acquire fences) ----
    __threadfence();
    __syncthreads();
    if (threadIdx.x == 0) {
      atomicAdd(cnt, 1u);
      u32 target = (u32)(t + 1) * (u32)NBLK;
      while (__hip_atomic_load(cnt, __ATOMIC_RELAXED, __HIP_MEMORY_SCOPE_AGENT) < target)
        __builtin_amdgcn_s_sleep(2);
    }
    __syncthreads();
    __threadfence();
  }
}

// ---------- workspace layout (bytes) ----------
#define CNT_OFF  0u
#define XB_OFF   256u
#define XB_SZ    ((size_t)T_STEPS * BATCH * IDIM * 2)         // 32 MB
#define WXT_OFF  (XB_OFF + XB_SZ)
#define WXT_SZ   ((size_t)4096 * IDIM * 2)                    // 4 MB
#define WHT_OFF  (WXT_OFF + WXT_SZ)
#define WHT_SZ   ((size_t)4096 * HDIM * 2)                    // 8 MB
#define B_OFF    (WHT_OFF + WHT_SZ)
#define B_SZ     ((size_t)4096 * 4)
#define H0_OFF   (B_OFF + B_SZ)
#define H_SZ     ((size_t)BATCH * HDIM * 2)
#define H1_OFF   (H0_OFF + H_SZ)

extern "C" void kernel_launch(void* const* d_in, const int* in_sizes, int n_in,
                              void* d_out, int out_size, void* d_ws, size_t ws_size,
                              hipStream_t stream) {
  const float* x   = (const float*)d_in[0];
  const float* h0  = (const float*)d_in[1];
  const float* c0  = (const float*)d_in[2];
  const float* Wii = (const float*)d_in[3];
  const float* Whi = (const float*)d_in[4];
  const float* bi  = (const float*)d_in[5];
  const float* Wif = (const float*)d_in[6];
  const float* Whf = (const float*)d_in[7];
  const float* bf_ = (const float*)d_in[8];
  const float* Wio = (const float*)d_in[9];
  const float* Who = (const float*)d_in[10];
  const float* bo  = (const float*)d_in[11];
  const float* Wig = (const float*)d_in[12];
  const float* Whg = (const float*)d_in[13];
  const float* bg  = (const float*)d_in[14];

  char* ws = (char*)d_ws;
  u32*  cnt  = (u32*)(ws + CNT_OFF);
  u16*  xb   = (u16*)(ws + XB_OFF);
  u16*  wxt  = (u16*)(ws + WXT_OFF);
  u16*  wht  = (u16*)(ws + WHT_OFF);
  float* bcat = (float*)(ws + B_OFF);
  u16*  hb0  = (u16*)(ws + H0_OFF);
  u16*  hb1  = (u16*)(ws + H1_OFF);
  float* out = (float*)d_out;

  // reset grid barrier counter (deterministic per launch, capture-safe)
  k_zero_cnt<<<1, 32, 0, stream>>>(cnt);

  // x -> bf16
  {
    int n = T_STEPS * BATCH * IDIM;
    k_cvt_bf16<<<(n + 255) / 256, 256, 0, stream>>>(x, xb, n);
  }
  // h0 -> bf16 ping buffer
  {
    int n = BATCH * HDIM;
    k_cvt_bf16<<<(n + 255) / 256, 256, 0, stream>>>(h0, hb0, n);
  }
  // transposed bf16 weights, gate order i,f,g,o
  {
    int n = 4096 * IDIM;
    k_pack_wT<<<(n + 255) / 256, 256, 0, stream>>>(Wii, Wif, Wig, Wio, wxt, IDIM);
  }
  {
    int n = 4096 * HDIM;
    k_pack_wT<<<(n + 255) / 256, 256, 0, stream>>>(Whi, Whf, Whg, Who, wht, HDIM);
  }
  k_pack_bias<<<16, 256, 0, stream>>>(bi, bf_, bg, bo, bcat);

  // persistent recurrent scan
  k_lstm_scan<<<NBLK, BLKTHR, 0, stream>>>(xb, wxt, wht, bcat, c0, hb0, hb1, out, cnt);

  (void)in_sizes; (void)n_in; (void)out_size; (void)ws_size;
}
